// LSTM_Seq2Dis_15272903704706
// MI455X (gfx1250) — matrix operand order repaired and checked by reading, not compiled, not run on hardware
//
#include <hip/hip_runtime.h>

typedef __attribute__((ext_vector_type(16))) _Float16 v16h;
typedef __attribute__((ext_vector_type(8)))  float    v8f;

// Under-aligned vector loads (emb rows are only 8B aligned; DWORD-align mode
// only requires 4B for b128/b64 vector loads on CDNA5).
typedef float f32x4u __attribute__((vector_size(16), aligned(4)));
typedef float f32x2u __attribute__((vector_size(8),  aligned(4)));

#define B_SZ   2048
#define T_SZ   256
#define E_SZ   50
#define H_SZ   50
#define G4     200   // 4*H

#define LOG2E_F 1.4426950408889634f

// Branch-free native transcendentals: v_exp_f32 (base-2) + v_rcp_f32.
__device__ __forceinline__ float fast_sigmoid(float x) {
    float e = __builtin_amdgcn_exp2f(x * -LOG2E_F);         // exp(-x)
    return __builtin_amdgcn_rcpf(1.f + e);                  // 1/(1+exp(-x))
}
__device__ __forceinline__ float fast_tanh(float x) {
    float u = __builtin_amdgcn_exp2f(x * (-2.f * LOG2E_F)); // exp(-2x)
    return fmaf(2.f, __builtin_amdgcn_rcpf(1.f + u), -1.f); // 2/(1+u) - 1
}

// ---- Fragment packing helpers (CDNA5 wave32 WMMA layouts, CDNA5 ISA §7.12.2) ----
// A-matrix 16x32 f16: lane l holds row M=l&15; entry e -> K = ko + (e&7) + ((e>>3)<<4) + ((l>>4)<<3)
__device__ __forceinline__ v16h packA_lds64(const _Float16* buf, int lane, int ko) {
    const int M  = lane & 15;
    const int kb = (lane >> 4) << 3;
    v16h a;
#pragma unroll
    for (int e = 0; e < 16; ++e) {
        int K = ko + (e & 7) + ((e >> 3) << 4) + kb;
        a[e] = buf[M * 64 + K];                 // buffer is [16][64], zero-padded
    }
    return a;
}

// x-part of A, ko=0: K runs kb+[0,8) and kb+[16,24); kb in {0,8} so K <= 31 < 50.
// Vectorized: 4x global_load_b128.
__device__ __forceinline__ v16h packA_x_lo(const float* row, int kb) {
    f32x4u a0 = *(const f32x4u*)(row + kb);
    f32x4u a1 = *(const f32x4u*)(row + kb + 4);
    f32x4u b0 = *(const f32x4u*)(row + kb + 16);
    f32x4u b1 = *(const f32x4u*)(row + kb + 20);
    v16h a;
#pragma unroll
    for (int e = 0; e < 4; ++e) {
        a[e]      = (_Float16)a0[e];
        a[e + 4]  = (_Float16)a1[e];
        a[e + 8]  = (_Float16)b0[e];
        a[e + 12] = (_Float16)b1[e];
    }
    return a;
}

// x-part of A, ko=32: K runs 32+kb+[0,8) (max 47 < 50, always valid) and
// 48+kb+[0,8) (only K=48,49 valid, i.e. kb==0, first two entries).
__device__ __forceinline__ v16h packA_x_hi(const float* row, int kb) {
    f32x4u a0 = *(const f32x4u*)(row + 32 + kb);
    f32x4u a1 = *(const f32x4u*)(row + 36 + kb);
    float t0 = 0.f, t1 = 0.f;
    if (kb == 0) {                              // half-wave-uniform branch
        f32x2u tt = *(const f32x2u*)(row + 48);
        t0 = tt[0]; t1 = tt[1];
    }
    v16h a;
#pragma unroll
    for (int e = 0; e < 4; ++e) {
        a[e]     = (_Float16)a0[e];
        a[e + 4] = (_Float16)a1[e];
    }
    a[8] = (_Float16)t0;
    a[9] = (_Float16)t1;
#pragma unroll
    for (int e = 10; e < 16; ++e) a[e] = (_Float16)0.f;
    return a;
}

// B-matrix 32x16 f16: lane l holds column N=l&15; entry e -> K = kbase + (e&7) + ((e>>3)<<4) + ((l>>4)<<3)
// Gate B fragment for padded tile (wave w, gate g): weight column = g*50 + (16w + N)
__device__ __forceinline__ v16h packB_gate(const float* Wk, const float* Uk,
                                           int lane, int kt, int g, int w) {
    const int c = w * 16 + (lane & 15);         // gate-local column (valid < 50)
    v16h bf;
#pragma unroll
    for (int e = 0; e < 16; ++e) {
        int K = kt * 32 + (e & 7) + ((e >> 3) << 4) + ((lane >> 4) << 3);   // 0..127 over [x | h]
        float v = 0.f;
        if (c < H_SZ) {
            if (K < E_SZ)                        v = Wk[K * G4 + g * H_SZ + c];
            else if (K >= 64 && (K - 64) < H_SZ) v = Uk[(K - 64) * G4 + g * H_SZ + c];
        }
        bf[e] = (_Float16)v;
    }
    return bf;
}

__device__ __forceinline__ v16h packB_w1(const float* W1, int lane, int kt, int w) {
    const int N = w * 16 + (lane & 15);
    v16h bf;
#pragma unroll
    for (int e = 0; e < 16; ++e) {
        int K = kt * 32 + (e & 7) + ((e >> 3) << 4) + ((lane >> 4) << 3);
        float v = (K < H_SZ && N < E_SZ) ? W1[K * E_SZ + N] : 0.f;
        bf[e] = (_Float16)v;
    }
    return bf;
}

// One block = 4 waves = one 16-row batch tile carried through all T timesteps.
__global__ __launch_bounds__(128) void lstm_seq2dis_fused(
    const int*   __restrict__ in_idx,  // [B,T]
    const float* __restrict__ emb,     // [VOCAB,E]
    const float* __restrict__ Wk,      // [E,4H]
    const float* __restrict__ Uk,      // [H,4H]
    const float* __restrict__ b,       // [4H]
    const float* __restrict__ W1,      // [H,E]
    const float* __restrict__ b1,      // [E]
    const float* __restrict__ W2,      // [E,1]
    const float* __restrict__ b2,      // [1]
    float*       __restrict__ out)     // [B,T]
{
    __shared__ _Float16 hbuf[16][64];  // h state, f16, zero-padded K
    __shared__ _Float16 dbuf[16][64];  // d1 then d2 staging
    __shared__ float    w2s[64];

    const int tid  = threadIdx.x;
    const int w    = tid >> 5;         // wave id 0..3 (owns gate columns 16w..16w+15)
    const int lane = tid & 31;
    const int b0   = blockIdx.x * 16;

    for (int i = tid; i < 16 * 64; i += 128) {
        (&hbuf[0][0])[i] = (_Float16)0.f;
        (&dbuf[0][0])[i] = (_Float16)0.f;
    }
    if (tid < 64) w2s[tid] = (tid < E_SZ) ? W2[tid] : 0.f;

    // Loop-invariant weight fragments live in VGPRs for all 256 steps.
    v16h wgf[4][4];                    // [gate][k-step] for concat [x|h] GEMM
#pragma unroll
    for (int g = 0; g < 4; ++g)
#pragma unroll
        for (int kt = 0; kt < 4; ++kt)
            wgf[g][kt] = packB_gate(Wk, Uk, lane, kt, g, w);
    v16h w1f[2];
    w1f[0] = packB_w1(W1, lane, 0, w);
    w1f[1] = packB_w1(W1, lane, 1, w);

    const int cgate = w * 16 + (lane & 15);     // this lane's gate-local / dense column
    float bias_g[4];
#pragma unroll
    for (int g = 0; g < 4; ++g) bias_g[g] = (cgate < H_SZ) ? b[g * H_SZ + cgate] : 0.f;
    const float bias1 = (cgate < E_SZ) ? b1[cgate] : 0.f;
    const float b2v   = b2[0];

    float creg[8];                              // LSTM cell state, rows rowbase..rowbase+7
#pragma unroll
    for (int r = 0; r < 8; ++r) creg[r] = 0.f;

    const int arow    = lane & 15;              // A-matrix row this lane feeds
    const int kb      = (lane >> 4) << 3;       // A-matrix K base for this lane
    const int rowbase = (lane >> 4) << 3;       // C-matrix row base for this lane

    __syncthreads();

    for (int t = 0; t < T_SZ; ++t) {
        const int idx_row = in_idx[(b0 + arow) * T_SZ + t];
        const float* row  = emb + idx_row * E_SZ;
        if (t + 1 < T_SZ) {                     // warm L2/L0 for next step's gather
            const int nidx = in_idx[(b0 + arow) * T_SZ + t + 1];
            __builtin_prefetch(emb + nidx * E_SZ, 0, 1);
        }

        // A = [x | h] : 16x128 f16 (x via wide b128 gathers, h from LDS)
        v16h af0 = packA_x_lo(row, kb);
        v16h af1 = packA_x_hi(row, kb);
        v16h af2 = packA_lds64(&hbuf[0][0], lane, 0);
        v16h af3 = packA_lds64(&hbuf[0][0], lane, 32);
        __syncthreads();                        // everyone done reading h(t-1)

        // z = [x|h] @ [Wk;Uk] + b  -- 4 gates x K=128, fp32 accumulate
        v8f accg[4];
#pragma unroll
        for (int g = 0; g < 4; ++g) {
            v8f acc;
#pragma unroll
            for (int r = 0; r < 8; ++r) acc[r] = bias_g[g];
            acc = __builtin_amdgcn_wmma_f32_16x16x32_f16(false, af0, false, wgf[g][0], (short)0, acc, false, false);
            acc = __builtin_amdgcn_wmma_f32_16x16x32_f16(false, af1, false, wgf[g][1], (short)0, acc, false, false);
            acc = __builtin_amdgcn_wmma_f32_16x16x32_f16(false, af2, false, wgf[g][2], (short)0, acc, false, false);
            acc = __builtin_amdgcn_wmma_f32_16x16x32_f16(false, af3, false, wgf[g][3], (short)0, acc, false, false);
            accg[g] = acc;
        }

        // Gate math entirely in registers, branch-free (i/f/g/o same lane, different accs)
#pragma unroll
        for (int r = 0; r < 8; ++r) {
            float iv = fast_sigmoid(accg[0][r]);
            float fv = fast_sigmoid(accg[1][r]);
            float gv = fast_tanh(accg[2][r]);
            float ov = fast_sigmoid(accg[3][r]);
            float cv = fmaf(fv, creg[r], iv * gv);
            creg[r]  = cv;
            float hv = ov * fast_tanh(cv);
            if (cgate < H_SZ) hbuf[rowbase + r][cgate] = (_Float16)hv;
        }
        __syncthreads();                        // h(t) visible to all waves

        // d1 = relu(h @ W1 + b1) : each wave computes one 16-col tile
        v16h a1lo = packA_lds64(&hbuf[0][0], lane, 0);
        v16h a1hi = packA_lds64(&hbuf[0][0], lane, 32);
        v8f acc1;
#pragma unroll
        for (int r = 0; r < 8; ++r) acc1[r] = bias1;
        acc1 = __builtin_amdgcn_wmma_f32_16x16x32_f16(false, a1lo, false, w1f[0], (short)0, acc1, false, false);
        acc1 = __builtin_amdgcn_wmma_f32_16x16x32_f16(false, a1hi, false, w1f[1], (short)0, acc1, false, false);
#pragma unroll
        for (int r = 0; r < 8; ++r)
            dbuf[rowbase + r][cgate] = (_Float16)fmaxf(acc1[r], 0.f);   // pad cols stay 0
        __syncthreads();                        // d1 ready

        // d2 = relu(d1 @ W1 + b1), reusing dbuf after all waves packed A
        v16h a2lo = packA_lds64(&dbuf[0][0], lane, 0);
        v16h a2hi = packA_lds64(&dbuf[0][0], lane, 32);
        __syncthreads();
        v8f acc2;
#pragma unroll
        for (int r = 0; r < 8; ++r) acc2[r] = bias1;
        acc2 = __builtin_amdgcn_wmma_f32_16x16x32_f16(false, a2lo, false, w1f[0], (short)0, acc2, false, false);
        acc2 = __builtin_amdgcn_wmma_f32_16x16x32_f16(false, a2hi, false, w1f[1], (short)0, acc2, false, false);
#pragma unroll
        for (int r = 0; r < 8; ++r)
            dbuf[rowbase + r][cgate] = (_Float16)fmaxf(acc2[r], 0.f);
        __syncthreads();                        // d2 ready

        // p = sigmoid(d2 @ W2 + b2) : one lane per batch row
        if (tid < 16) {
            float s = b2v;
#pragma unroll 10
            for (int cc = 0; cc < E_SZ; ++cc) s += (float)dbuf[tid][cc] * w2s[cc];
            out[(b0 + tid) * T_SZ + t] = fast_sigmoid(s);
        }
        __syncthreads();                        // protect dbuf/hbuf before next step
    }
}

extern "C" void kernel_launch(void* const* d_in, const int* in_sizes, int n_in,
                              void* d_out, int out_size, void* d_ws, size_t ws_size,
                              hipStream_t stream) {
    const int*   in_idx = (const int*)  d_in[0];
    const float* emb    = (const float*)d_in[1];
    const float* Wk     = (const float*)d_in[2];
    const float* Uk     = (const float*)d_in[3];
    const float* b      = (const float*)d_in[4];
    const float* W1     = (const float*)d_in[5];
    const float* b1     = (const float*)d_in[6];
    const float* W2     = (const float*)d_in[7];
    const float* b2     = (const float*)d_in[8];
    float* out = (float*)d_out;

    dim3 grid(B_SZ / 16);
    dim3 block(128);
    hipLaunchKernelGGL(lstm_seq2dis_fused, grid, block, 0, stream,
                       in_idx, emb, Wk, Uk, b, W1, b1, W2, b2, out);
}
